// LSTMCell_6347961663605
// MI455X (gfx1250) — matrix operand order repaired and checked by reading, not compiled, not run on hardware
//
#include <hip/hip_runtime.h>
#include <hip/hip_fp16.h>
#include <math.h>

typedef __attribute__((ext_vector_type(16))) _Float16 v16h;
typedef __attribute__((ext_vector_type(8)))  float    v8f;
typedef __attribute__((ext_vector_type(4)))  unsigned v4u;
typedef __attribute__((ext_vector_type(8)))  int      v8i;
typedef __attribute__((ext_vector_type(4)))  int      v4i;

#define B_DIM 4096
#define H_DIM 1024
#define K_DIM 2048   // I + H
#define NLV   0.05f

#if defined(__HIP_DEVICE_COMPILE__) && __has_builtin(__builtin_amdgcn_tensor_load_to_lds) && __has_builtin(__builtin_amdgcn_s_wait_tensorcnt)
#define HAVE_TDM 1
#else
#define HAVE_TDM 0
#endif

// ---- ws layout (bytes) ----
// [0,16)        : 4 ordered-uint max slots
// [256, 16640)  : bias_comb, 4096 floats
// [32768, +16MB): X fragments  (256 mtiles x 64 ktiles x 32 lanes x 16 f16)
// [+16MB, +32MB): W fragments  (256 ntiles x 64 ktiles x 32 lanes x 16 f16)
#define WS_BIAS  256
#define WS_XFRAG 32768
#define WS_WFRAG (32768 + 16u*1024u*1024u)

// ---------- helpers ----------
__device__ __forceinline__ unsigned orderedFloat(float f) {
    unsigned u = __float_as_uint(f);
    return (u & 0x80000000u) ? ~u : (u | 0x80000000u);
}
__device__ __forceinline__ float unorderFloat(unsigned u) {
    return __uint_as_float((u & 0x80000000u) ? (u & 0x7FFFFFFFu) : ~u);
}
__device__ __forceinline__ unsigned pcg_hash(unsigned x) {
    x = x * 747796405u + 2891336453u;
    unsigned w = ((x >> ((x >> 28u) + 4u)) ^ x) * 277803737u;
    return (w >> 22u) ^ w;
}
// deterministic standard normal from (index, salt) — Box-Muller
__device__ __forceinline__ float hash_normal(unsigned idx, unsigned salt) {
    unsigned h1 = pcg_hash(idx ^ (salt * 0x9E3779B9u));
    unsigned h2 = pcg_hash(h1 + 0x85EBCA6Bu);
    float u1 = (float)(h1 >> 8) * (1.0f / 16777216.0f) + 1.0e-7f;
    float u2 = (float)(h2 >> 8) * (1.0f / 16777216.0f);
    return sqrtf(-2.0f * __logf(u1)) * __cosf(6.283185307179586f * u2);
}
// quant_pass(x, 8, signed): clip [-127/128, 127/128], round-to-even on /128 grid
__device__ __forceinline__ float qs8(float x) {
    x = fminf(fmaxf(x, -0.9921875f), 0.9921875f);
    return rintf(x * 128.0f) * 0.0078125f;
}
// quant_pass(x, 8, unsigned): clip [0,1], round-to-even on /256 grid
__device__ __forceinline__ float qu8(float x) {
    x = fminf(fmaxf(x, 0.0f), 1.0f);
    return rintf(x * 256.0f) * 0.00390625f;
}
__device__ __forceinline__ float sigm(float x) { return 1.0f / (1.0f + __expf(-x)); }

// ---------- prep kernels ----------
__global__ void k_init_slots(unsigned* slots) {
    if (threadIdx.x < 4) slots[threadIdx.x] = 0u;
}

__global__ void k_reduce_max(const float* __restrict__ src, int n, unsigned* slot) {
    float m = -3.402823466e38f;
    for (int i = blockIdx.x * blockDim.x + threadIdx.x; i < n; i += gridDim.x * blockDim.x)
        m = fmaxf(m, src[i]);
    atomicMax(slot, orderedFloat(m));
}

// X = [q(input), q(hx)]  (4096 x 2048) -> WMMA A-fragment layout (f16)
__global__ void k_xfrag(const float* __restrict__ input, const float* __restrict__ hx,
                        _Float16* __restrict__ xfrag) {
    int t = blockIdx.x * blockDim.x + threadIdx.x;
    int j = t & 15, lane = (t >> 4) & 31, kt = (t >> 9) & 63, mt = t >> 15;
    int row  = mt * 16 + (lane & 15);
    int half = lane >> 4;
    int kj   = (j < 8) ? (j + 8 * half) : (16 + (j - 8) + 8 * half);
    int k    = kt * 32 + kj;
    float x  = (k < H_DIM) ? input[row * H_DIM + k] : hx[row * H_DIM + (k - H_DIM)];
    xfrag[t] = (_Float16)qs8(x);
}

// W[k][n] = weightT + noise  (2048 x 4096) -> WMMA B-fragment layout (f16)
__global__ void k_wfrag(const float* __restrict__ w_ih, const float* __restrict__ w_hh,
                        const unsigned* __restrict__ slots, _Float16* __restrict__ wfrag) {
    int t = blockIdx.x * blockDim.x + threadIdx.x;
    int j = t & 15, lane = (t >> 4) & 31, kt = (t >> 9) & 63, nt = t >> 15;
    int n = nt * 16 + (lane & 15);
    int k = kt * 32 + ((j < 8) ? (j + 8 * (lane >> 4)) : (16 + (j - 8) + 8 * (lane >> 4)));
    float w;
    if (k < H_DIM) {
        float mx = unorderFloat(slots[0]);
        w = w_ih[n * H_DIM + k] + hash_normal((unsigned)(k * 4096 + n), 1u) * mx * NLV;
    } else {
        int kk = k - H_DIM;
        float mx = unorderFloat(slots[1]);
        w = w_hh[n * H_DIM + kk] + hash_normal((unsigned)(kk * 4096 + n), 2u) * mx * NLV;
    }
    wfrag[t] = (_Float16)w;
}

__global__ void k_bias(const float* __restrict__ b_ih, const float* __restrict__ b_hh,
                       const unsigned* __restrict__ slots, float* __restrict__ bias_comb) {
    int n = blockIdx.x * blockDim.x + threadIdx.x;
    if (n >= 4096) return;
    float m_ih = unorderFloat(slots[2]);
    float m_hh = unorderFloat(slots[3]);
    bias_comb[n] = b_ih[n] + b_hh[n]
                 + hash_normal((unsigned)n, 3u) * m_ih * NLV
                 + hash_normal((unsigned)n, 4u) * m_hh * NLV;
}

// ---------- TDM stage load ----------
// One descriptor moves 32KB: 4 rows (gates) x 8KB (8 K-step fragments),
// global row stride 4MB (gate stride in wfrag), packed contiguously into LDS.
// D# per ISA ch.8: group0 {count=1, lds_addr, global_addr, type=2},
// group1 {data_size=8B, tensor_dim0=1024u, tensor_dim1=4, tile 1024x4, dim0_stride=524288u}.
#if HAVE_TDM
__device__ __forceinline__ void tdm_issue_stage(const char* wfrag_base, int hblock, int s) {
    unsigned long long ga = (unsigned long long)(uintptr_t)wfrag_base
                          + (unsigned long long)hblock * 65536ull
                          + (unsigned long long)s * 8192ull;
    unsigned lds_off = (unsigned)((s & 1) * 32768);  // dynamic LDS is the only LDS -> offset 0
    v4u g0;
    g0.x = 1u;                                        // count=1 (valid user descriptor)
    g0.y = lds_off;                                   // lds_addr
    g0.z = (unsigned)(ga & 0xFFFFFFFFull);            // global_addr[31:0]
    g0.w = (unsigned)(ga >> 32) | (2u << 30);         // global_addr[56:32] | type=2
    v8i g1;
    g1[0] = 0x30000;                                  // data_size=3 (8 bytes/unit)
    g1[1] = (int)(1024u << 16);                       // tensor_dim0 = 1024 units
    g1[2] = (int)(4u << 16);                          // tensor_dim1 = 4
    g1[3] = (int)(1024u << 16);                       // tile_dim0 = 1024 units (8KB row)
    g1[4] = 4;                                        // tile_dim1 = 4 rows
    g1[5] = 524288;                                   // tensor_dim0_stride = 4MB / 8B
    g1[6] = 0;
    g1[7] = 0;
    v4i z4 = {0, 0, 0, 0};
    v8i z8 = {0, 0, 0, 0, 0, 0, 0, 0};
    // clang-23 lane: 6-arg form (extra int32x8 descriptor words, unused for 2D tile)
    __builtin_amdgcn_tensor_load_to_lds(g0, g1, z4, z4, z8, 0);
}
#endif

// ---------- fused WMMA GEMM + LSTM epilogue ----------
// grid: 16 mblocks * 64 hblocks = 1024 WGs; 256 threads = 8 waves.
// WG tile: M=256 (8 waves x 2 mtiles), N=16 columns for all 4 gates.
// B staged in 32KB chunks (8 K-steps) via TDM, double buffered (64KB LDS),
// one s_wait_tensorcnt + two barriers per stage. A is register-pipelined.
__global__ __launch_bounds__(256)
void k_lstm_gemm(const _Float16* __restrict__ xfrag, const _Float16* __restrict__ wfrag,
                 const float* __restrict__ bias, const float* __restrict__ cx,
                 float* __restrict__ hy, float* __restrict__ cy) {
    extern __shared__ char ldsB[];            // 2 stages x 32768 bytes
    const int tid    = threadIdx.x;
    const int lane   = tid & 31;
    const int wave   = tid >> 5;
    const int mblock = blockIdx.x >> 6;       // 0..15
    const int hblock = blockIdx.x & 63;       // 0..63

    const int mt0 = mblock * 16 + wave * 2;
    const v16h* __restrict__ A0 = ((const v16h*)xfrag) + (size_t)(mt0 * 64) * 32 + lane;
    const v16h* __restrict__ A1 = A0 + (size_t)64 * 32;

    v8f acc[8] = {};                          // [m*4 + gate]

#if HAVE_TDM
    if (wave == 0) tdm_issue_stage((const char*)wfrag, hblock, 0);
#else
    // fallback: cooperative copy of stage 0 (128B per thread)
    {
        const char* src = (const char*)wfrag + (size_t)hblock * 65536;
        for (int i = 0; i < 8; ++i) {
            int ofs = tid * 16 + i * 4096;    // [0,32768): g = ofs>>13, rest linear
            int g = ofs >> 13, r = ofs & 8191;
            *(float4*)(ldsB + ofs) = *(const float4*)(src + (size_t)g * 4194304 + r);
        }
    }
#endif

    v16h a0c = A0[0];
    v16h a1c = A1[0];

    for (int s = 0; s < 8; ++s) {
#if HAVE_TDM
        if (wave == 0 && s + 1 < 8) tdm_issue_stage((const char*)wfrag, hblock, s + 1);
        if (s < 7) __builtin_amdgcn_s_wait_tensorcnt(1);   // oldest (stage s) complete
        else       __builtin_amdgcn_s_wait_tensorcnt(0);
        __syncthreads();
#else
        float4 stg[8];
        if (s + 1 < 8) {
            const char* src = (const char*)wfrag + (size_t)hblock * 65536 + (size_t)(s + 1) * 8192;
            for (int i = 0; i < 8; ++i) {
                int ofs = tid * 16 + i * 4096;
                int g = ofs >> 13, r = ofs & 8191;
                stg[i] = *(const float4*)(src + (size_t)g * 4194304 + r);
            }
        }
        __syncthreads();   // stage s visible
#endif

        const char* Bs = ldsB + (s & 1) * 32768;
#pragma unroll
        for (int k8 = 0; k8 < 8; ++k8) {
            const int kt = s * 8 + k8;
            v16h a0n, a1n;
            const bool more = (kt + 1 < 64);
            if (more) {
                a0n = A0[(kt + 1) * 32];
                a1n = A1[(kt + 1) * 32];
            }
            const v16h* Bv = (const v16h*)(Bs + k8 * 1024);   // gate stride 8192B = 256 v16h
            v16h b0 = Bv[lane];
            v16h b1 = Bv[256 + lane];
            v16h b2 = Bv[512 + lane];
            v16h b3 = Bv[768 + lane];

            acc[0] = __builtin_amdgcn_wmma_f32_16x16x32_f16(false, a0c, false, b0, (short)0, acc[0], false, false);
            acc[1] = __builtin_amdgcn_wmma_f32_16x16x32_f16(false, a0c, false, b1, (short)0, acc[1], false, false);
            acc[2] = __builtin_amdgcn_wmma_f32_16x16x32_f16(false, a0c, false, b2, (short)0, acc[2], false, false);
            acc[3] = __builtin_amdgcn_wmma_f32_16x16x32_f16(false, a0c, false, b3, (short)0, acc[3], false, false);
            acc[4] = __builtin_amdgcn_wmma_f32_16x16x32_f16(false, a1c, false, b0, (short)0, acc[4], false, false);
            acc[5] = __builtin_amdgcn_wmma_f32_16x16x32_f16(false, a1c, false, b1, (short)0, acc[5], false, false);
            acc[6] = __builtin_amdgcn_wmma_f32_16x16x32_f16(false, a1c, false, b2, (short)0, acc[6], false, false);
            acc[7] = __builtin_amdgcn_wmma_f32_16x16x32_f16(false, a1c, false, b3, (short)0, acc[7], false, false);

            if (more) { a0c = a0n; a1c = a1n; }
        }

#if HAVE_TDM
        __syncthreads();   // all reads of stage s done before TDM refills it next iter
#else
        if (s + 1 < 8) {
            const int dst = ((s + 1) & 1) * 32768;
            for (int i = 0; i < 8; ++i)
                *(float4*)(ldsB + dst + tid * 16 + i * 4096) = stg[i];
        }
        __syncthreads();
#endif
    }

    // ---- epilogue: C/D layout -> VGPR v, lane half: row = v + 8*half, col = lane&15
    const int n_local = lane & 15;
    const int half    = lane >> 4;
    const int h       = hblock * 16 + n_local;
    const float b_i = bias[h];
    const float b_f = bias[1024 + h];
    const float b_g = bias[2048 + h];
    const float b_o = bias[3072 + h];

#pragma unroll
    for (int m = 0; m < 2; ++m) {
        const int Mbase = (mt0 + m) * 16;
#pragma unroll
        for (int v = 0; v < 8; ++v) {
            int row = Mbase + v + 8 * half;
            float gi = acc[m * 4 + 0][v] + b_i;
            float gf = acc[m * 4 + 1][v] + b_f;
            float gc = acc[m * 4 + 2][v] + b_g;
            float go = acc[m * 4 + 3][v] + b_o;
            float ig = qu8(sigm(gi));
            float fg = qu8(sigm(gf));
            float cg = qs8(tanhf(gc));
            float og = qu8(sigm(go));
            float cv = cx[row * H_DIM + h];
            float c2 = qs8((qs8(fg * cv) + qs8(ig * cg)) * 0.5f);   // /CY_DIV
            float h2 = qs8(og * qs8(tanhf(c2 * 2.0f)));             // *CY_SCALE
            hy[row * H_DIM + h] = h2;
            cy[row * H_DIM + h] = c2;
        }
    }
}

extern "C" void kernel_launch(void* const* d_in, const int* in_sizes, int n_in,
                              void* d_out, int out_size, void* d_ws, size_t ws_size,
                              hipStream_t stream) {
    const float* input = (const float*)d_in[0];
    const float* hx    = (const float*)d_in[1];
    const float* cx    = (const float*)d_in[2];
    const float* w_ih  = (const float*)d_in[3];
    const float* w_hh  = (const float*)d_in[4];
    const float* b_ih  = (const float*)d_in[5];
    const float* b_hh  = (const float*)d_in[6];
    (void)in_sizes; (void)n_in; (void)out_size; (void)ws_size;

    char* ws = (char*)d_ws;
    unsigned* slots     = (unsigned*)ws;
    float*    bias_comb = (float*)(ws + WS_BIAS);
    _Float16* xfrag     = (_Float16*)(ws + WS_XFRAG);
    _Float16* wfrag     = (_Float16*)(ws + WS_WFRAG);

    float* hy = (float*)d_out;                        // [4096,1024]
    float* cy = (float*)d_out + (size_t)B_DIM * H_DIM;

    k_init_slots<<<1, 32, 0, stream>>>(slots);
    k_reduce_max<<<256, 256, 0, stream>>>(w_ih, 4096 * 1024, slots + 0);
    k_reduce_max<<<256, 256, 0, stream>>>(w_hh, 4096 * 1024, slots + 1);
    k_reduce_max<<<16, 256, 0, stream>>>(b_ih, 4096, slots + 2);
    k_reduce_max<<<16, 256, 0, stream>>>(b_hh, 4096, slots + 3);

    k_xfrag<<<32768, 256, 0, stream>>>(input, hx, xfrag);
    k_wfrag<<<32768, 256, 0, stream>>>(w_ih, w_hh, slots, wfrag);
    k_bias<<<16, 256, 0, stream>>>(b_ih, b_hh, slots, bias_comb);

    k_lstm_gemm<<<16 * 64, 256, 2 * 32768, stream>>>(xfrag, wfrag, bias_comb, cx, hy, cy);
}
